// ConvCapsule_52939766890499
// MI455X (gfx1250) — compile-verified, matrix operand order checked by reading, not run on hardware
//
#include <hip/hip_runtime.h>

typedef __attribute__((ext_vector_type(16))) _Float16 v16h;
typedef __attribute__((ext_vector_type(8)))  _Float16 v8h;
typedef __attribute__((ext_vector_type(8)))  float    v8f;

// ---------------------------------------------------------------------------
// Kernel 0: pack W [3,3,16,256] f32 (== row-major [144,256]) into a
// transposed, K-padded f16 matrix Wt[ncol][160].  With the wave32 WMMA B
// layout, a lane's 8-element half-group is then one contiguous b128 load.
// ---------------------------------------------------------------------------
__global__ __launch_bounds__(256) void pack_w_f16(
    const float* __restrict__ Wm,    // [144,256]
    _Float16* __restrict__ Wt)       // [256,160], kk >= 144 zero-padded
{
    const int idx  = blockIdx.x * 256 + threadIdx.x;   // 0 .. 256*160-1
    const int ncol = idx / 160;
    const int kk   = idx - ncol * 160;
    Wt[idx] = (_Float16)((kk < 144) ? Wm[kk * 256 + ncol] : 0.0f);
}

// ---------------------------------------------------------------------------
// Kernel 1: conv as im2col + WMMA GEMM.
//   votes[k][p][n] = sum_{kh,kw,cin} x[k%8, h+kh-1, w+kw-1, k/8, cin]*W[kh,kw,cin,n]
//   M = 16 pixels/tile, N = 256 (2 n-tiles per wave, 8 waves), K = 144 pad 160.
//   A half-groups = 8 consecutive cin at one (kh,kw) tap -> two float4 loads.
//   B half-groups = contiguous f16 b128 loads from Wt.
// ---------------------------------------------------------------------------
__global__ __launch_bounds__(256) void conv_votes_wmma(
    const float* __restrict__ x,       // [8,32,32,8,16]
    const _Float16* __restrict__ Wt,   // [256,160]
    float* __restrict__ votes)         // [64,1024,256] in d_ws
{
    const int k     = blockIdx.x;        // conv batch 0..63
    const int mtile = blockIdx.y;        // 0..63, 16 pixels each
    const int wave  = threadIdx.x >> 5;  // 0..7
    const int lane  = threadIdx.x & 31;
    const int m     = lane & 15;         // A row / C column within tile
    const int hi    = lane >> 4;         // upper-half-of-wave flag

    const int bx   = k & 7;              // x batch index  (= k % 8)
    const int ninx = k >> 3;             // x nin index    (= k / 8)
    const int p0   = mtile * 16;

    // pixel whose im2col row this lane contributes to A
    const int p = p0 + m;
    const int h = p >> 5;
    const int w = p & 31;

    // gfx1250 prefetch of this lane's center-tap row (global_prefetch_b8)
    __builtin_prefetch(x + (((bx * 32 + h) * 32 + w) * 8 + ninx) * 16, 0, 3);

    // ---- Build A fragments for all 5 K-tiles (ISA 16-bit A 16x32 layout) ----
    // element i<8  -> kk = kt*32 +      hi*8 + i   (half-group g=0)
    // element i>=8 -> kk = kt*32 + 16 + hi*8 + (i-8) (half-group g=1)
    v16h afrag[5];
#pragma unroll
    for (int kt = 0; kt < 5; ++kt) {
        v16h a;
#pragma unroll
        for (int g = 0; g < 2; ++g) {
            const int kk0 = kt * 32 + g * 16 + hi * 8;  // 8 consecutive kk
            float f[8];
#pragma unroll
            for (int j = 0; j < 8; ++j) f[j] = 0.0f;
            if (kk0 < 144) {
                const int kh   = kk0 / 48;
                const int rem  = kk0 - kh * 48;
                const int kw   = rem >> 4;
                const int cin0 = rem & 15;              // 0 or 8
                const int hh   = h + kh - 1;
                const int ww   = w + kw - 1;
                if (hh >= 0 && hh < 32 && ww >= 0 && ww < 32) {
                    const float* xp =
                        x + (((bx * 32 + hh) * 32 + ww) * 8 + ninx) * 16 + cin0;
                    const float4 f0 = *(const float4*)xp;        // 16B aligned
                    const float4 f1 = *(const float4*)(xp + 4);
                    f[0] = f0.x; f[1] = f0.y; f[2] = f0.z; f[3] = f0.w;
                    f[4] = f1.x; f[5] = f1.y; f[6] = f1.z; f[7] = f1.w;
                }
            }
#pragma unroll
            for (int j = 0; j < 8; ++j) a[g * 8 + j] = (_Float16)f[j];
        }
        afrag[kt] = a;
    }

    // ---- Each wave computes two 16x16 C tiles along N ----
#pragma unroll
    for (int nt2 = 0; nt2 < 2; ++nt2) {
        const int nt   = wave * 2 + nt2;
        const int ncol = nt * 16 + m;            // this lane's B/C column
        const _Float16* Wrow = Wt + ncol * 160;
        v8f acc = {};
#pragma unroll
        for (int kt = 0; kt < 5; ++kt) {
            const v8h blo = *(const v8h*)(Wrow + kt * 32 + hi * 8);       // b128
            const v8h bhi = *(const v8h*)(Wrow + kt * 32 + 16 + hi * 8);  // b128
            v16h bfr;
#pragma unroll
            for (int j = 0; j < 8; ++j) { bfr[j] = blo[j]; bfr[8 + j] = bhi[j]; }
            // D = A x B + C, fp32 accumulate
            acc = __builtin_amdgcn_wmma_f32_16x16x32_f16(
                false, afrag[kt], false, bfr, (short)0, acc, false, false);
        }
        // C/D layout: lane column = ncol; VGPR r -> row r + hi*8
#pragma unroll
        for (int r = 0; r < 8; ++r) {
            const int pp = p0 + r + hi * 8;
            votes[(k * 1024 + pp) * 256 + ncol] = acc[r];
        }
    }
}

// ---------------------------------------------------------------------------
// Kernel 2: fused 2-iteration dynamic routing. One block per output pixel,
// thread t = nc*16 + dc. All 8 Nin votes live in registers; squash /
// agreement via 16-lane shfl_xor reductions; softmax over NC via tiny LDS.
// ---------------------------------------------------------------------------
__global__ __launch_bounds__(256) void routing_kernel(
    const float* __restrict__ votes,  // [64,1024,256]
    const float* __restrict__ bias,   // [16,16]
    float* __restrict__ out)          // [8,32,32,16,16]
{
    const int pix = blockIdx.x;       // b*1024 + p
    const int b   = pix >> 10;
    const int p   = pix & 1023;
    const int t   = threadIdx.x;      // = nc*16 + dc
    const int nc  = t >> 4;
    const int dc  = t & 15;

    float v[8];
#pragma unroll
    for (int nin = 0; nin < 8; ++nin) {
        const int k = b * 8 + nin;    // reference's reshape quirk, reproduced
        v[nin] = votes[(k * 1024 + p) * 256 + t];
    }
    const float bv = bias[t];

    __shared__ float lgs[8][16];

    // ---- routing iteration 1: logits==0 -> route = 1/16 uniformly ----
    float preact = 0.0f;
#pragma unroll
    for (int nin = 0; nin < 8; ++nin) preact += v[nin];
    preact = preact * (1.0f / 16.0f) + bv;

    // squash: reduce preact^2 over the 16 dc lanes (half-wave)
    float sq = preact * preact;
    sq += __shfl_xor(sq, 1, 32);
    sq += __shfl_xor(sq, 2, 32);
    sq += __shfl_xor(sq, 4, 32);
    sq += __shfl_xor(sq, 8, 32);
    const float act1 = preact * (sq / ((1.0f + sq) * sqrtf(sq + 1e-9f)));

    // agreement: logits[nin, nc] = sum_dc v*act
    float lg[8];
#pragma unroll
    for (int nin = 0; nin < 8; ++nin) {
        float d = v[nin] * act1;
        d += __shfl_xor(d, 1, 32);
        d += __shfl_xor(d, 2, 32);
        d += __shfl_xor(d, 4, 32);
        d += __shfl_xor(d, 8, 32);
        lg[nin] = d;
        if (dc == 0) lgs[nin][nc] = d;
    }
    __syncthreads();

    // ---- routing iteration 2: route = softmax over nc, per nin ----
    float pre2 = bv;
#pragma unroll
    for (int nin = 0; nin < 8; ++nin) {
        float mx = -3.0e38f;
#pragma unroll
        for (int j = 0; j < 16; ++j) mx = fmaxf(mx, lgs[nin][j]);
        float den = 0.0f;
#pragma unroll
        for (int j = 0; j < 16; ++j) den += __expf(lgs[nin][j] - mx);
        const float route = __expf(lg[nin] - mx) / den;
        pre2 += route * v[nin];
    }
    float sq2 = pre2 * pre2;
    sq2 += __shfl_xor(sq2, 1, 32);
    sq2 += __shfl_xor(sq2, 2, 32);
    sq2 += __shfl_xor(sq2, 4, 32);
    sq2 += __shfl_xor(sq2, 8, 32);
    const float act2 = pre2 * (sq2 / ((1.0f + sq2) * sqrtf(sq2 + 1e-9f)));

    out[pix * 256 + t] = act2;        // [B,H,W,NC,DC] row-major
}

// ---------------------------------------------------------------------------
extern "C" void kernel_launch(void* const* d_in, const int* in_sizes, int n_in,
                              void* d_out, int out_size, void* d_ws, size_t ws_size,
                              hipStream_t stream) {
    const float* x  = (const float*)d_in[0];   // [8,32,32,8,16]
    const float* Wm = (const float*)d_in[1];   // [3,3,16,256] == row-major [144,256]
    const float* bb = (const float*)d_in[2];   // [1,1,16,16]
    float* out = (float*)d_out;                // [8,32,32,16,16]

    // scratch layout: votes f32 [64*1024*256] (64 MB), then Wt f16 [256*160]
    float*    votes = (float*)d_ws;
    _Float16* Wt    = (_Float16*)((char*)d_ws + (size_t)64 * 1024 * 1024);

    pack_w_f16<<<160, 256, 0, stream>>>(Wm, Wt);
    dim3 g1(64, 64);
    conv_votes_wmma<<<g1, 256, 0, stream>>>(x, Wt, votes);
    routing_kernel<<<8192, 256, 0, stream>>>(votes, bb, out);
}
